// MultiScaleLinearAttention_77197742178867
// MI455X (gfx1250) — matrix operand order repaired and checked by reading, not compiled, not run on hardware
//
#include <hip/hip_runtime.h>
#include <cstdint>
#include <cstddef>

// ---------------- types / helpers ----------------
typedef __attribute__((ext_vector_type(16))) _Float16 v16h;
typedef __attribute__((ext_vector_type(8)))  float    v8f;

union FragU { v16h h; uint4 q[2]; };

__device__ inline v8f v8f_zero() {
  v8f z;
#pragma unroll
  for (int i = 0; i < 8; ++i) z[i] = 0.0f;
  return z;
}

__device__ inline float waveSum(float v) {
#pragma unroll
  for (int off = 16; off; off >>= 1) v += __shfl_xor(v, off, 32);
  return v;
}

// ---------------- generic f16 WMMA GEMM:  C = A(MxK) * W(NxK)^T + bias ----------------
// wave: 64x64 tile (4x4 wmma), block: 128 threads = 2x2 waves = 128x128 tile.
// Requires: M % 128 == 0, Ncol % 128 == 0, K % 32 == 0.
__global__ void __launch_bounds__(128)
gemm_f16_wmma(const _Float16* __restrict__ A, const _Float16* __restrict__ W,
              const float* __restrict__ bias, float* __restrict__ Cout,
              int M, int Ncol, int K) {
  const int lane = threadIdx.x & 31;
  const int wave = threadIdx.x >> 5;
  const int m0 = blockIdx.y * 128 + (wave & 1) * 64;
  const int n0 = blockIdx.x * 128 + (wave >> 1) * 64;

  const int arow = lane & 15;
  const int akb  = (lane & 16) ? 8 : 0;    // A: K sub-base 0 or 8 (+ runs at +16)
  const int bcol = lane & 15;
  const int bkb  = (lane & 16) ? 8 : 0;    // B: K sub-base 0 or 8 (+ runs at +16)

  v8f acc[4][4];
#pragma unroll
  for (int i = 0; i < 4; ++i)
#pragma unroll
    for (int j = 0; j < 4; ++j) acc[i][j] = v8f_zero();

  const _Float16* Abase = A + (long long)m0 * K;
  const _Float16* Wbase = W + (long long)n0 * K;

  for (int k0 = 0; k0 < K; k0 += 32) {
    FragU a[4], b[4];
#pragma unroll
    for (int mi = 0; mi < 4; ++mi) {
      const _Float16* p = Abase + (long long)(mi * 16 + arow) * K + k0 + akb;
      a[mi].q[0] = *(const uint4*)p;
      a[mi].q[1] = *(const uint4*)(p + 16);
      __builtin_prefetch(p + 32, 0, 1);
    }
#pragma unroll
    for (int ni = 0; ni < 4; ++ni) {
      const _Float16* p = Wbase + (long long)(ni * 16 + bcol) * K + k0 + bkb;
      b[ni].q[0] = *(const uint4*)p;
      b[ni].q[1] = *(const uint4*)(p + 16);
    }
#pragma unroll
    for (int mi = 0; mi < 4; ++mi)
#pragma unroll
      for (int ni = 0; ni < 4; ++ni)
        acc[mi][ni] = __builtin_amdgcn_wmma_f32_16x16x32_f16(
            false, a[mi].h, false, b[ni].h, (short)0, acc[mi][ni], false, false);
  }

  const int ccol = lane & 15;
  const int rofs = (lane & 16) ? 8 : 0;
#pragma unroll
  for (int ni = 0; ni < 4; ++ni) {
    const int cc = n0 + ni * 16 + ccol;
    const float bv = bias ? bias[cc] : 0.0f;
#pragma unroll
    for (int mi = 0; mi < 4; ++mi) {
#pragma unroll
      for (int r = 0; r < 8; ++r) {
        const int rr = m0 + mi * 16 + rofs + r;
        Cout[(long long)rr * Ncol + cc] = acc[mi][ni][r] + bv;
      }
    }
  }
}

// ---------------- elementwise / staging kernels ----------------
__global__ void cvt_f16_kernel(const float* __restrict__ in, _Float16* __restrict__ out, long long n) {
  long long i = (long long)blockIdx.x * blockDim.x + threadIdx.x;
  if (i < n) out[i] = (_Float16)in[i];
}

__global__ void softplus_kernel(const float* __restrict__ s, float* __restrict__ sp) {
  int c = threadIdx.x;
  float v = s[c];
  sp[c] = (v > 20.0f) ? v : log1pf(expf(v));
}

// im2col for conv1: 4x4 stride 4 -> rows = B*256, K = 4096 (k = c*16 + dy*4 + dx)
__global__ void im2col1_kernel(const float* __restrict__ x, _Float16* __restrict__ out) {
  int idx = blockIdx.x * blockDim.x + threadIdx.x;           // (row, c)
  if (idx >= 16 * 256 * 256) return;
  int c = idx & 255, row = idx >> 8;
  int b = row >> 8, pos = row & 255, oy = pos >> 4, ox = pos & 15;
  long long obase = (long long)row * 4096 + c * 16;
#pragma unroll
  for (int t = 0; t < 16; ++t) {
    int dy = t >> 2, dx = t & 3;
    int n = (oy * 4 + dy) * 64 + ox * 4 + dx;
    out[obase + t] = (_Float16)x[((long long)(b * 4096 + n)) * 256 + c];
  }
}

// im2col for conv2: 2x2 stride 2 -> rows = B*1024, K = 1024 (k = c*4 + dy*2 + dx)
__global__ void im2col2_kernel(const float* __restrict__ x, _Float16* __restrict__ out) {
  int idx = blockIdx.x * blockDim.x + threadIdx.x;
  if (idx >= 16 * 1024 * 256) return;
  int c = idx & 255, row = idx >> 8;
  int b = row >> 10, pos = row & 1023, oy = pos >> 5, ox = pos & 31;
  long long obase = (long long)row * 1024 + c * 4;
#pragma unroll
  for (int t = 0; t < 4; ++t) {
    int dy = t >> 1, dx = t & 1;
    int n = (oy * 2 + dy) * 64 + ox * 2 + dx;
    out[obase + t] = (_Float16)x[((long long)(b * 4096 + n)) * 256 + c];
  }
}

// LayerNorm + exact GELU, one wave per row of 256, f16 output
__global__ void __launch_bounds__(256)
ln_gelu_kernel(const float* __restrict__ in, const float* __restrict__ g,
               const float* __restrict__ bp, _Float16* __restrict__ out, int rows) {
  int row = blockIdx.x * (blockDim.x >> 5) + (threadIdx.x >> 5);
  if (row >= rows) return;
  int lane = threadIdx.x & 31;
  long long base = (long long)row * 256;
  float v[8]; float s = 0.0f;
#pragma unroll
  for (int i = 0; i < 8; ++i) { v[i] = in[base + lane + 32 * i]; s += v[i]; }
  float mu = waveSum(s) * (1.0f / 256.0f);
  float s2 = 0.0f;
#pragma unroll
  for (int i = 0; i < 8; ++i) { float d = v[i] - mu; s2 += d * d; }
  float rinv = rsqrtf(waveSum(s2) * (1.0f / 256.0f) + 1e-5f);
#pragma unroll
  for (int i = 0; i < 8; ++i) {
    int c = lane + 32 * i;
    float y = (v[i] - mu) * rinv * g[c] + bp[c];
    float ge = 0.5f * y * (1.0f + erff(y * 0.70710678118f));
    out[base + c] = (_Float16)ge;
  }
}

// q path: gather scrambled row from qflat(B,N,C), relu+eps, /sp, pow3 + renorm
__global__ void __launch_bounds__(256)
qnorm_kernel(const float* __restrict__ qflat, const float* __restrict__ sp,
             float* __restrict__ qn, int rows) {
  int row = blockIdx.x * (blockDim.x >> 5) + (threadIdx.x >> 5);   // b*4096 + m
  if (row >= rows) return;
  int lane = threadIdx.x & 31;
  int b = row >> 12, m = row & 4095;
  int h = m >> 9;
  float v[8]; float ss = 0.0f;
#pragma unroll
  for (int i = 0; i < 8; ++i) {
    int c2 = lane + 32 * i;
    int n = ((m & 511) << 3) + (c2 >> 5);
    int d = c2 & 31;
    float xv = qflat[((long long)(b * 4096 + n)) * 256 + h * 32 + d];
    xv = fmaxf(xv, 0.0f) + 1e-6f;
    xv = xv / sp[c2];
    v[i] = xv; ss += xv * xv;
  }
  float n1 = sqrtf(waveSum(ss));
  float s3 = 0.0f;
#pragma unroll
  for (int i = 0; i < 8; ++i) { float t = v[i] * v[i] * v[i]; v[i] = t; s3 += t * t; }
  float inv = n1 / sqrtf(waveSum(s3));
  long long ob = (long long)row * 256;
#pragma unroll
  for (int i = 0; i < 8; ++i) qn[ob + lane + 32 * i] = v[i] * inv;
}

// k path: k + pe, relu+eps, /sp, pow3 + renorm (natural layout)
__global__ void __launch_bounds__(256)
knorm_kernel(const float* __restrict__ k, const float* __restrict__ pe,
             const float* __restrict__ sp, float* __restrict__ kn, int rows, int n_kv) {
  int row = blockIdx.x * (blockDim.x >> 5) + (threadIdx.x >> 5);
  if (row >= rows) return;
  int lane = threadIdx.x & 31;
  int j = row % n_kv;
  long long base = (long long)row * 256;
  float v[8]; float ss = 0.0f;
#pragma unroll
  for (int i = 0; i < 8; ++i) {
    int c = lane + 32 * i;
    float xv = k[base + c] + pe[(long long)j * 256 + c];
    xv = fmaxf(xv, 0.0f) + 1e-6f;
    xv = xv / sp[c];
    v[i] = xv; ss += xv * xv;
  }
  float n1 = sqrtf(waveSum(ss));
  float s3 = 0.0f;
#pragma unroll
  for (int i = 0; i < 8; ++i) { float t = v[i] * v[i] * v[i]; v[i] = t; s3 += t * t; }
  float inv = n1 / sqrtf(waveSum(s3));
#pragma unroll
  for (int i = 0; i < 8; ++i) kn[base + lane + 32 * i] = v[i] * inv;
}

// y = (x@Wkv^T) (B,h,h,2C) -> transpose(0,3,1,2) -> reshape -> k,v (B,n_kv,C)
__global__ void kv_scramble_kernel(const float* __restrict__ t, float* __restrict__ k,
                                   float* __restrict__ v, int n_kv, int h1) {
  int idx = blockIdx.x * blockDim.x + threadIdx.x;   // (b, n, ch)
  if (idx >= 16 * n_kv * 256) return;
  int ch = idx & 255, rest = idx >> 8;
  int n = rest % n_kv, b = rest / n_kv;
  int kf = n * 512 + ch;
  int c = kf / n_kv, rem = kf % n_kv, i = rem / h1, j = rem % h1;
  k[idx] = t[((long long)(b * n_kv) + i * h1 + j) * 512 + c];
  int vf = kf + 256;
  c = vf / n_kv; rem = vf % n_kv; i = rem / h1; j = rem % h1;
  v[idx] = t[((long long)(b * n_kv) + i * h1 + j) * 512 + c];
}

// vr[b,j',c'] = v(B,H,n_kv,D).reshape(B, n_kv, C)
__global__ void vr_scramble_kernel(const float* __restrict__ v, float* __restrict__ vr, int n_kv) {
  int idx = blockIdx.x * blockDim.x + threadIdx.x;
  if (idx >= 16 * n_kv * 256) return;
  int cp = idx & 255, rest = idx >> 8;
  int jp = rest % n_kv, b = rest / n_kv;
  int g8 = n_kv >> 3;
  int h = jp / g8;
  int j = (jp % g8) * 8 + (cp >> 5);
  int d = cp & 31;
  vr[idx] = v[((long long)(b * n_kv) + j) * 256 + h * 32 + d];
}

// km[b, h*32+d] = mean_j kn[b, j, h*32+d]
__global__ void km_kernel(const float* __restrict__ kn, float* __restrict__ km, int n_kv) {
  int idx = blockIdx.x * blockDim.x + threadIdx.x;   // 16*256
  if (idx >= 16 * 256) return;
  int b = idx >> 8, c = idx & 255;
  float s = 0.0f;
  for (int j = 0; j < n_kv; ++j) s += kn[((long long)(b * n_kv) + j) * 256 + c];
  km[idx] = s / (float)n_kv;
}

// kvm[(b*8+h)*1024 + d*32 + e] = (1/4096) * sum_j kn[b,j,h*32+d] * v[b,j,h*32+e]
__global__ void __launch_bounds__(256)
kvmat_kernel(const float* __restrict__ kn, const float* __restrict__ v,
             float* __restrict__ kvm, int n_kv) {
  __shared__ float lk[256], lv[256];
  int b = blockIdx.x >> 3, h = blockIdx.x & 7;
  int tid = threadIdx.x;
  int e = tid & 31, d0 = (tid >> 5) * 4;
  float a0 = 0, a1 = 0, a2 = 0, a3 = 0;
  for (int jc = 0; jc < n_kv; jc += 8) {
    int jl = tid >> 5, dd = tid & 31;
    long long base = ((long long)(b * n_kv) + jc + jl) * 256 + h * 32 + dd;
    lk[tid] = kn[base];
    lv[tid] = v[base];
    __syncthreads();
#pragma unroll
    for (int q = 0; q < 8; ++q) {
      float vv = lv[q * 32 + e];
      a0 += lk[q * 32 + d0 + 0] * vv;
      a1 += lk[q * 32 + d0 + 1] * vv;
      a2 += lk[q * 32 + d0 + 2] * vv;
      a3 += lk[q * 32 + d0 + 3] * vv;
    }
    __syncthreads();
  }
  const float s = 1.0f / 4096.0f;
  long long ob = (long long)blockIdx.x * 1024;
  kvm[ob + (d0 + 0) * 32 + e] = a0 * s;
  kvm[ob + (d0 + 1) * 32 + e] = a1 * s;
  kvm[ob + (d0 + 2) * 32 + e] = a2 * s;
  kvm[ob + (d0 + 3) * 32 + e] = a3 * s;
}

// Fused epilogue: out = (q @ kv) * z + depthwise5x5(linear-interp(vr))
// One block per (b, n); thread c = h*32 + e. The dwc channel split (g = c>>5,
// ch = c&31) coincides exactly with (h, e), so both terms share one thread.
// Writes f16 into concat buffer column block [scaleOff, scaleOff+256).
__global__ void __launch_bounds__(256)
att_dwc_out_kernel(const float* __restrict__ qn, const float* __restrict__ km,
                   const float* __restrict__ kvm, const float* __restrict__ vr,
                   const float* __restrict__ wdwc, const float* __restrict__ bdwc,
                   _Float16* __restrict__ cat, int n_kv, int scaleOff) {
  int bn = blockIdx.x;            // b*4096 + n
  int b = bn >> 12;
  int n = bn & 4095;
  int c = threadIdx.x;
  int h = c >> 5, e = c & 31;

  // ---- linear attention term ----
  float qv = qn[(long long)bn * 256 + c];
  float zs = waveSum(qv * km[b * 256 + c]);
  float z = 1.0f / (zs + 1e-6f);
  const float* kvmh = kvm + (long long)(b * 8 + h) * 1024;
  float s = 0.0f;
#pragma unroll
  for (int d = 0; d < 32; ++d) {
    float qd = __shfl(qv, d, 32);
    s += qd * kvmh[d * 32 + e];
  }

  // ---- interp + depthwise 5x5 conv term (g = h, ch = e) ----
  int y = n >> 6, x = n & 63;
  float acc = bdwc[e];
  float Lf = (float)n_kv, ratio = Lf / 4096.0f;
  long long vb = (long long)b * n_kv * 256;
#pragma unroll
  for (int dy = -2; dy <= 2; ++dy) {
    int yy = y + dy; if (yy < 0 || yy >= 64) continue;
#pragma unroll
    for (int dx = -2; dx <= 2; ++dx) {
      int xx = x + dx; if (xx < 0 || xx >= 64) continue;
      int nn = h * 512 + yy * 8 + (xx >> 3);
      int cc = (xx & 7) * 32 + e;
      float pos = ((float)nn + 0.5f) * ratio - 0.5f;
      pos = fminf(fmaxf(pos, 0.0f), Lf - 1.0f);
      int i0 = (int)pos;
      int i1 = (i0 + 1 < n_kv) ? i0 + 1 : n_kv - 1;
      float w = pos - (float)i0;
      float val = vr[vb + (long long)i0 * 256 + cc] * (1.0f - w) +
                  vr[vb + (long long)i1 * 256 + cc] * w;
      acc += wdwc[e * 25 + (dy + 2) * 5 + (dx + 2)] * val;
    }
  }

  float o = s * z + acc;
  cat[(long long)bn * 512 + scaleOff + c] = (_Float16)o;
}

// ---------------- workspace layout (bytes) ----------------
static constexpr size_t SZ_BIG16 = (size_t)16 * 4096 * 256 * 2;   // 32MB: x_f16 / im2col1 / im2col2
static constexpr size_t SZ_QNS   = (size_t)16 * 4096 * 256 * 4;   // 64MB
static constexpr size_t SZ_TMP   = (size_t)16 * 4096 * 256 * 4;   // 64MB: qflat / conv / kv outs
static constexpr size_t SZ_X1F   = (size_t)4096 * 256 * 2;
static constexpr size_t SZ_X2F   = (size_t)16384 * 256 * 2;
static constexpr size_t SZ_WQ    = (size_t)256 * 256 * 2;
static constexpr size_t SZ_WSR1  = (size_t)256 * 4096 * 2;
static constexpr size_t SZ_WSR2  = (size_t)256 * 1024 * 2;
static constexpr size_t SZ_WKV   = (size_t)512 * 256 * 2;
static constexpr size_t SZ_WPROJ = (size_t)256 * 512 * 2;
static constexpr size_t SZ_SP    = 1024;
static constexpr size_t SZ_S1    = (size_t)16 * 256 * 256 * 4;    // 4MB each (k1/v1/kn1/vr1)
static constexpr size_t SZ_S2    = (size_t)16 * 1024 * 256 * 4;   // 16MB each
static constexpr size_t SZ_KM    = (size_t)16 * 256 * 4;
static constexpr size_t SZ_KVM   = (size_t)16 * 8 * 32 * 32 * 4;
static constexpr size_t SZ_CAT   = (size_t)16 * 4096 * 512 * 2;   // 64MB

static constexpr size_t O_X16   = 0;
static constexpr size_t O_QNS   = O_X16   + SZ_BIG16;
static constexpr size_t O_TMP   = O_QNS   + SZ_QNS;
static constexpr size_t O_X1F   = O_TMP   + SZ_TMP;
static constexpr size_t O_X2F   = O_X1F   + SZ_X1F;
static constexpr size_t O_WQ    = O_X2F   + SZ_X2F;
static constexpr size_t O_WSR1  = O_WQ    + SZ_WQ;
static constexpr size_t O_WSR2  = O_WSR1  + SZ_WSR1;
static constexpr size_t O_WKV1  = O_WSR2  + SZ_WSR2;
static constexpr size_t O_WKV2  = O_WKV1  + SZ_WKV;
static constexpr size_t O_WPROJ = O_WKV2  + SZ_WKV;
static constexpr size_t O_SP    = O_WPROJ + SZ_WPROJ;
static constexpr size_t O_K1    = O_SP    + SZ_SP;
static constexpr size_t O_V1    = O_K1    + SZ_S1;
static constexpr size_t O_KN1   = O_V1    + SZ_S1;
static constexpr size_t O_VR1   = O_KN1   + SZ_S1;
static constexpr size_t O_K2    = O_VR1   + SZ_S1;
static constexpr size_t O_V2    = O_K2    + SZ_S2;
static constexpr size_t O_KN2   = O_V2    + SZ_S2;
static constexpr size_t O_VR2   = O_KN2   + SZ_S2;
static constexpr size_t O_KM    = O_VR2   + SZ_S2;
static constexpr size_t O_KVM   = O_KM    + SZ_KM;
static constexpr size_t O_CAT   = O_KVM   + SZ_KVM;

extern "C" void kernel_launch(void* const* d_in, const int* in_sizes, int n_in,
                              void* d_out, int out_size, void* d_ws, size_t ws_size,
                              hipStream_t stream) {
  (void)in_sizes; (void)n_in; (void)out_size; (void)ws_size;
  const float* x     = (const float*)d_in[0];
  const float* Wq    = (const float*)d_in[1];
  const float* bq    = (const float*)d_in[2];
  const float* Wsr1  = (const float*)d_in[3];
  const float* bsr1  = (const float*)d_in[4];
  const float* g1    = (const float*)d_in[5];
  const float* b1    = (const float*)d_in[6];
  const float* Wsr2  = (const float*)d_in[7];
  const float* bsr2  = (const float*)d_in[8];
  const float* g2    = (const float*)d_in[9];
  const float* b2    = (const float*)d_in[10];
  const float* Wkv1  = (const float*)d_in[11];
  const float* bkv1  = (const float*)d_in[12];
  const float* Wkv2  = (const float*)d_in[13];
  const float* bkv2  = (const float*)d_in[14];
  const float* pe1   = (const float*)d_in[15];
  const float* pe2   = (const float*)d_in[16];
  const float* Wproj = (const float*)d_in[17];
  const float* bproj = (const float*)d_in[18];
  const float* Wdwc  = (const float*)d_in[19];
  const float* bdwc  = (const float*)d_in[20];
  const float* scale = (const float*)d_in[21];
  float* out = (float*)d_out;

  char* ws = (char*)d_ws;
  _Float16* big16  = (_Float16*)(ws + O_X16);     // x_f16, then im2col1, then im2col2
  float*    qns    = (float*)(ws + O_QNS);
  float*    tmp    = (float*)(ws + O_TMP);        // qflat / conv / kv outs
  _Float16* x1f    = (_Float16*)(ws + O_X1F);
  _Float16* x2f    = (_Float16*)(ws + O_X2F);
  _Float16* wq16   = (_Float16*)(ws + O_WQ);
  _Float16* wsr116 = (_Float16*)(ws + O_WSR1);
  _Float16* wsr216 = (_Float16*)(ws + O_WSR2);
  _Float16* wkv116 = (_Float16*)(ws + O_WKV1);
  _Float16* wkv216 = (_Float16*)(ws + O_WKV2);
  _Float16* wpr16  = (_Float16*)(ws + O_WPROJ);
  float*    sp     = (float*)(ws + O_SP);
  float*    k1     = (float*)(ws + O_K1);
  float*    v1     = (float*)(ws + O_V1);
  float*    kn1    = (float*)(ws + O_KN1);
  float*    vr1    = (float*)(ws + O_VR1);
  float*    k2     = (float*)(ws + O_K2);
  float*    v2     = (float*)(ws + O_V2);
  float*    kn2    = (float*)(ws + O_KN2);
  float*    vr2    = (float*)(ws + O_VR2);
  float*    km     = (float*)(ws + O_KM);
  float*    kvm    = (float*)(ws + O_KVM);
  _Float16* cat16  = (_Float16*)(ws + O_CAT);

  const long long NX = 16LL * 4096 * 256;
  // --- stage f16 copies ---
  cvt_f16_kernel<<<(unsigned)((NX + 255) / 256), 256, 0, stream>>>(x, big16, NX);
  cvt_f16_kernel<<<(256 * 256 + 255) / 256, 256, 0, stream>>>(Wq, wq16, 256 * 256);
  cvt_f16_kernel<<<(256 * 4096 + 255) / 256, 256, 0, stream>>>(Wsr1, wsr116, 256 * 4096);
  cvt_f16_kernel<<<(256 * 1024 + 255) / 256, 256, 0, stream>>>(Wsr2, wsr216, 256 * 1024);
  cvt_f16_kernel<<<(512 * 256 + 255) / 256, 256, 0, stream>>>(Wkv1, wkv116, 512 * 256);
  cvt_f16_kernel<<<(512 * 256 + 255) / 256, 256, 0, stream>>>(Wkv2, wkv216, 512 * 256);
  cvt_f16_kernel<<<(256 * 512 + 255) / 256, 256, 0, stream>>>(Wproj, wpr16, 256 * 512);
  softplus_kernel<<<1, 256, 0, stream>>>(scale, sp);

  // --- q = x @ Wq^T + bq ; then scrambled power-normalization ---
  gemm_f16_wmma<<<dim3(2, 512), 128, 0, stream>>>(big16, wq16, bq, tmp, 65536, 256, 256);
  qnorm_kernel<<<65536 / 8, 256, 0, stream>>>(tmp, sp, qns, 65536);

  // --- scale 1: conv1 (4x4 s4) -> LN+GELU -> kv1 -> scramble -> knorm ---
  im2col1_kernel<<<(16 * 256 * 256 + 255) / 256, 256, 0, stream>>>(x, big16);
  gemm_f16_wmma<<<dim3(2, 32), 128, 0, stream>>>(big16, wsr116, bsr1, tmp, 4096, 256, 4096);
  ln_gelu_kernel<<<4096 / 8, 256, 0, stream>>>(tmp, g1, b1, x1f, 4096);
  gemm_f16_wmma<<<dim3(4, 32), 128, 0, stream>>>(x1f, wkv116, bkv1, tmp, 4096, 512, 256);
  kv_scramble_kernel<<<(16 * 256 * 256 + 255) / 256, 256, 0, stream>>>(tmp, k1, v1, 256, 16);
  vr_scramble_kernel<<<(16 * 256 * 256 + 255) / 256, 256, 0, stream>>>(v1, vr1, 256);
  knorm_kernel<<<4096 / 8, 256, 0, stream>>>(k1, pe1, sp, kn1, 4096, 256);

  // --- scale 2: conv2 (2x2 s2) -> LN+GELU -> kv2 -> scramble -> knorm ---
  im2col2_kernel<<<(16 * 1024 * 256 + 255) / 256, 256, 0, stream>>>(x, big16);
  gemm_f16_wmma<<<dim3(2, 128), 128, 0, stream>>>(big16, wsr216, bsr2, tmp, 16384, 256, 1024);
  ln_gelu_kernel<<<16384 / 8, 256, 0, stream>>>(tmp, g2, b2, x2f, 16384);
  gemm_f16_wmma<<<dim3(4, 128), 128, 0, stream>>>(x2f, wkv216, bkv2, tmp, 16384, 512, 256);
  kv_scramble_kernel<<<(16 * 1024 * 256 + 255) / 256, 256, 0, stream>>>(tmp, k2, v2, 1024, 32);
  vr_scramble_kernel<<<(16 * 1024 * 256 + 255) / 256, 256, 0, stream>>>(v2, vr2, 1024);
  knorm_kernel<<<16384 / 8, 256, 0, stream>>>(k2, pe2, sp, kn2, 16384, 1024);

  // --- linear attention + fused interp/dwc epilogue, scale 1 ---
  km_kernel<<<(16 * 256 + 255) / 256, 256, 0, stream>>>(kn1, km, 256);
  kvmat_kernel<<<128, 256, 0, stream>>>(kn1, v1, kvm, 256);
  att_dwc_out_kernel<<<65536, 256, 0, stream>>>(qns, km, kvm, vr1, Wdwc, bdwc, cat16, 256, 0);

  // --- linear attention + fused interp/dwc epilogue, scale 2 ---
  km_kernel<<<(16 * 256 + 255) / 256, 256, 0, stream>>>(kn2, km, 1024);
  kvmat_kernel<<<128, 256, 0, stream>>>(kn2, v2, kvm, 1024);
  att_dwc_out_kernel<<<65536, 256, 0, stream>>>(qns, km, kvm, vr2, Wdwc, bdwc, cat16, 1024, 256);

  // --- final projection: out = cat @ Wproj^T + bproj ---
  gemm_f16_wmma<<<dim3(2, 512), 128, 0, stream>>>(cat16, wpr16, bproj, out, 65536, 256, 512);
}
